// QKVAttentionLegacy_77592879169589
// MI455X (gfx1250) — matrix operand order repaired and checked by reading, not compiled, not run on hardware
//
#include <hip/hip_runtime.h>
#include <hip/hip_bf16.h>

typedef __attribute__((ext_vector_type(16))) __bf16        v16bf;
typedef __attribute__((ext_vector_type(2)))  __bf16        v2bf;
typedef __attribute__((ext_vector_type(8)))  float         v8f;
typedef __attribute__((ext_vector_type(8)))  unsigned int  v8u;
typedef unsigned short u16;
typedef unsigned int   u32;

// qkv (4, 3072, 2048) fp32; 16 heads -> 64 batch-heads, ch = 64, L = 2048.
// out (4, 1024, 2048) fp32.
#define CH     64
#define LEN    2048
#define BM     64          // query rows per block
#define BN     64          // key cols per inner tile
#define QS     72          // LDS row stride (shorts), 16B-aligned, conflict pad
#define SS     68          // score row stride (floats)

// Pack two f32 -> two bf16 (RNE) in one instruction when available.
#if __has_builtin(__builtin_amdgcn_cvt_pk_bf16_f32)
__device__ __forceinline__ u32 pk_bf16(float a, float b) {
    v2bf r = __builtin_amdgcn_cvt_pk_bf16_f32(a, b);
    return __builtin_bit_cast(u32, r);
}
#else
__device__ __forceinline__ u32 pk_bf16(float a, float b) {
    v2bf r;
    r[0] = (__bf16)a;            // clang lowers to v_cvt bf16 (RNE)
    r[1] = (__bf16)b;
    return __builtin_bit_cast(u32, r);
}
#endif

// Build a 16xbf16 WMMA fragment from two 16-byte LDS reads.
__device__ __forceinline__ v16bf ldfrag2(const u16* p0, const u16* p1) {
    uint4 lo = *reinterpret_cast<const uint4*>(p0);
    uint4 hi = *reinterpret_cast<const uint4*>(p1);
    v8u u;
    u[0] = lo.x; u[1] = lo.y; u[2] = lo.z; u[3] = lo.w;
    u[4] = hi.x; u[5] = hi.y; u[6] = hi.z; u[7] = hi.w;
    return __builtin_bit_cast(v16bf, u);
}

__device__ __forceinline__ v8f wmma_bf16(v16bf a, v16bf b, v8f c) {
    return __builtin_amdgcn_wmma_f32_16x16x32_bf16(
        false, a, false, b, (short)0, c, false, false);
}

__global__ void __launch_bounds__(256)
qkv_attn_flash_kernel(const float* __restrict__ qkv, float* __restrict__ out) {
    __shared__ __align__(16) u16   q_s[BM][QS];   // Q  [t][c]  * ch^-1/4
    __shared__ __align__(16) u16   k_s[BN][QS];   // K^T [s][c] * ch^-1/4
    __shared__ __align__(16) u16   v_s[CH][QS];   // V  [c][s]
    __shared__ __align__(16) u16   p_s[BM][QS];   // softmax probs [t][s]
    __shared__ __align__(16) float sc_s[BM][SS];  // raw scores f32
    __shared__ float mrow[BM], lrow[BM], crow[BM];

    const int tid  = threadIdx.x;
    const int wave = tid >> 5;
    const int lane = tid & 31;
    const int lm   = lane & 15;     // row (A/C) or col (B/C) index
    const int lg   = lane >> 4;     // half-wave select

    const int head = blockIdx.x >> 5;        // 64 batch-heads
    const int t0   = (blockIdx.x & 31) * BM; // query tile origin

    // wave -> two owned 16x16 tiles of the 64x64 grid (same row-block mt)
    const int mt  = wave >> 1;
    const int nt0 = (wave & 1) * 2;
    const int nt1 = nt0 + 1;

    const float scale = 0.3535533905932738f; // ch^-0.25, ch = 64
    const float* __restrict__ qb = qkv + (size_t)head * 192 * LEN;   // q rows 0..63
    const float* __restrict__ kb = qb + (size_t)64 * LEN;            // k rows
    const float* __restrict__ vb = qb + (size_t)128 * LEN;           // v rows

    // per-thread staging coordinates: 4 float4 quads each for K and V tiles
    int sc_c[4], sc_s4[4];
    #pragma unroll
    for (int j = 0; j < 4; ++j) {
        int idx = tid + 256 * j;           // 0..1023 quads
        sc_c[j]  = idx >> 4;               // channel row 0..63
        sc_s4[j] = (idx & 15) * 4;         // s (or t) offset, multiple of 4
    }

    // ---- stage Q tile (64 t x 64 ch): float4 loads, pk-cvt, transpose ----
    #pragma unroll
    for (int j = 0; j < 4; ++j) {
        const int c = sc_c[j], t4 = sc_s4[j];
        float4 qv = *reinterpret_cast<const float4*>(&qb[(size_t)c * LEN + t0 + t4]);
        u32 p01 = pk_bf16(qv.x * scale, qv.y * scale);
        u32 p23 = pk_bf16(qv.z * scale, qv.w * scale);
        q_s[t4 + 0][c] = (u16)p01;
        q_s[t4 + 1][c] = (u16)(p01 >> 16);
        q_s[t4 + 2][c] = (u16)p23;
        q_s[t4 + 3][c] = (u16)(p23 >> 16);
    }
    if (tid < BM) { mrow[tid] = -3.0e38f; lrow[tid] = 0.0f; }

    v8f o0 = {}; v8f o1 = {};   // O accumulators, live across the whole loop

    // ---- software pipeline: preload tile 0 K/V into registers ----
    float4 kr[4], vr[4];
    #pragma unroll
    for (int j = 0; j < 4; ++j) {
        kr[j] = *reinterpret_cast<const float4*>(&kb[(size_t)sc_c[j] * LEN + sc_s4[j]]);
        vr[j] = *reinterpret_cast<const float4*>(&vb[(size_t)sc_c[j] * LEN + sc_s4[j]]);
    }

    for (int s0 = 0; s0 < LEN; s0 += BN) {
        __syncthreads();   // previous PV finished with k_s/v_s/p_s
        // commit registers (tile s0) to LDS: K transposed, V packed b64 stores
        #pragma unroll
        for (int j = 0; j < 4; ++j) {
            const int c = sc_c[j], s4 = sc_s4[j];
            u32 k01 = pk_bf16(kr[j].x * scale, kr[j].y * scale);
            u32 k23 = pk_bf16(kr[j].z * scale, kr[j].w * scale);
            k_s[s4 + 0][c] = (u16)k01;
            k_s[s4 + 1][c] = (u16)(k01 >> 16);
            k_s[s4 + 2][c] = (u16)k23;
            k_s[s4 + 3][c] = (u16)(k23 >> 16);
            uint2 pv;
            pv.x = pk_bf16(vr[j].x, vr[j].y);
            pv.y = pk_bf16(vr[j].z, vr[j].w);
            *reinterpret_cast<uint2*>(&v_s[c][s4]) = pv;
        }
        // issue next tile's global loads now; latency overlaps QK/softmax/PV
        if (s0 + BN < LEN) {
            #pragma unroll
            for (int j = 0; j < 4; ++j) {
                kr[j] = *reinterpret_cast<const float4*>(
                    &kb[(size_t)sc_c[j] * LEN + (s0 + BN) + sc_s4[j]]);
                vr[j] = *reinterpret_cast<const float4*>(
                    &vb[(size_t)sc_c[j] * LEN + (s0 + BN) + sc_s4[j]]);
            }
        }
        __syncthreads();

        // ---- S = Q K^T : two 16x16 tiles per wave, K-dim 64 = 2 x 32 ----
        const u16* aq = &q_s[mt * 16 + lm][0];
        #pragma unroll
        for (int ti = 0; ti < 2; ++ti) {
            int nt = nt0 + ti;
            const u16* bk = &k_s[nt * 16 + lm][0];
            v8f acc = {};
            v16bf a = ldfrag2(aq + 8 * lg,      aq + 8 * lg + 16);
            v16bf b = ldfrag2(bk + 8 * lg,      bk + 8 * lg + 16);
            acc = wmma_bf16(a, b, acc);
            a = ldfrag2(aq + 32 + 8 * lg,  aq + 32 + 8 * lg + 16);
            b = ldfrag2(bk + 32 + 8 * lg,  bk + 32 + 8 * lg + 16);
            acc = wmma_bf16(a, b, acc);
            #pragma unroll
            for (int r = 0; r < 8; ++r)
                sc_s[mt * 16 + 8 * lg + r][nt * 16 + lm] = acc[r];
        }
        __syncthreads();

        // ---- online softmax: 4 adjacent lanes per row, shuffle reductions ----
        const int trow = tid >> 2;
        const int tq   = tid & 3;
        const int sb   = tq * 16;
        {
            float pmax = sc_s[trow][sb];
            #pragma unroll
            for (int s = 1; s < 16; ++s) pmax = fmaxf(pmax, sc_s[trow][sb + s]);
            // reduce across the 4 lanes of this row (same wave, lanes tq^1, tq^2)
            pmax = fmaxf(pmax, __shfl_xor(pmax, 1, 32));
            float rmax = fmaxf(pmax, __shfl_xor(pmax, 2, 32));

            // NOTE: tq==0 may update mrow[trow] while siblings read it; benign,
            // since fmax(mnew, rmax) == mnew when mnew = fmax(mprev, rmax).
            const float mprev = mrow[trow];
            float mnew = fmaxf(mprev, rmax);
            float sum = 0.0f;
            #pragma unroll
            for (int s = 0; s < 16; s += 2) {
                float p0 = __expf(sc_s[trow][sb + s]     - mnew);
                float p1 = __expf(sc_s[trow][sb + s + 1] - mnew);
                sum += p0 + p1;
                *reinterpret_cast<u32*>(&p_s[trow][sb + s]) = pk_bf16(p0, p1);
            }
            sum += __shfl_xor(sum, 1, 32);
            sum += __shfl_xor(sum, 2, 32);
            if (tq == 0) {
                float corr = __expf(mprev - mnew);
                mrow[trow] = mnew;
                crow[trow] = corr;
                lrow[trow] = lrow[trow] * corr + sum;
            }
        }
        __syncthreads();   // crow / p_s visible to all waves

        // ---- rescale O, then O += P V^T ----
        #pragma unroll
        for (int r = 0; r < 8; ++r) {
            float cf = crow[mt * 16 + 8 * lg + r];
            o0[r] *= cf; o1[r] *= cf;
        }
        const u16* ap = &p_s[mt * 16 + lm][0];
        const u16* b0 = &v_s[nt0 * 16 + lm][0];
        const u16* b1 = &v_s[nt1 * 16 + lm][0];
        v16bf a  = ldfrag2(ap + 8 * lg,       ap + 8 * lg + 16);
        v16bf f0 = ldfrag2(b0 + 8 * lg,       b0 + 8 * lg + 16);
        v16bf f1 = ldfrag2(b1 + 8 * lg,       b1 + 8 * lg + 16);
        o0 = wmma_bf16(a, f0, o0);
        o1 = wmma_bf16(a, f1, o1);
        a  = ldfrag2(ap + 32 + 8 * lg,   ap + 32 + 8 * lg + 16);
        f0 = ldfrag2(b0 + 32 + 8 * lg,   b0 + 32 + 8 * lg + 16);
        f1 = ldfrag2(b1 + 32 + 8 * lg,   b1 + 32 + 8 * lg + 16);
        o0 = wmma_bf16(a, f0, o0);
        o1 = wmma_bf16(a, f1, o1);
    }
    __syncthreads();  // lrow final

    // ---- normalize by row sums, vectorized float4 stores to out[c][t] ----
    float linv[8];
    #pragma unroll
    for (int r = 0; r < 8; ++r)
        linv[r] = 1.0f / lrow[mt * 16 + 8 * lg + r];

    float* __restrict__ ob = out + (size_t)head * CH * LEN;
    const int trow0 = t0 + mt * 16 + 8 * lg;
    {
        int c = nt0 * 16 + lm;
        float4 lo = make_float4(o0[0] * linv[0], o0[1] * linv[1],
                                o0[2] * linv[2], o0[3] * linv[3]);
        float4 hi = make_float4(o0[4] * linv[4], o0[5] * linv[5],
                                o0[6] * linv[6], o0[7] * linv[7]);
        *reinterpret_cast<float4*>(&ob[(size_t)c * LEN + trow0])     = lo;
        *reinterpret_cast<float4*>(&ob[(size_t)c * LEN + trow0 + 4]) = hi;
        c = nt1 * 16 + lm;
        lo = make_float4(o1[0] * linv[0], o1[1] * linv[1],
                         o1[2] * linv[2], o1[3] * linv[3]);
        hi = make_float4(o1[4] * linv[4], o1[5] * linv[5],
                         o1[6] * linv[6], o1[7] * linv[7]);
        *reinterpret_cast<float4*>(&ob[(size_t)c * LEN + trow0])     = lo;
        *reinterpret_cast<float4*>(&ob[(size_t)c * LEN + trow0 + 4]) = hi;
    }
}

extern "C" void kernel_launch(void* const* d_in, const int* in_sizes, int n_in,
                              void* d_out, int out_size, void* d_ws, size_t ws_size,
                              hipStream_t stream) {
    (void)in_sizes; (void)n_in; (void)out_size; (void)d_ws; (void)ws_size;
    const float* qkv = (const float*)d_in[0];
    float* out = (float*)d_out;
    // 64 batch-heads x 32 query tiles of 64 rows
    qkv_attn_flash_kernel<<<dim3(2048), dim3(256), 0, stream>>>(qkv, out);
}
